// FastAttention_30356828848812
// MI455X (gfx1250) — matrix operand order repaired and checked by reading, not compiled, not run on hardware
//
#include <hip/hip_runtime.h>
#include <hip/hip_bf16.h>

// ---------------------------------------------------------------------------
// CDNA5 (gfx1250) wave32 BF16-WMMA pipeline for linear ("fast") attention.
//   out = (Q' (K'^T V)) Wout + bout      (associativity: no softmax present)
// All operands pre-converted to bf16 once; all GEMMs run on
// v_wmma_f32_16x16x32_bf16; A-tiles staged to LDS by the Tensor Data Mover.
// ---------------------------------------------------------------------------

typedef __bf16 bf16_t;
typedef bf16_t v16bf __attribute__((ext_vector_type(16)));
typedef float  v8f   __attribute__((ext_vector_type(8)));
typedef unsigned int u32x4 __attribute__((ext_vector_type(4)));
typedef int i32x4 __attribute__((ext_vector_type(4)));
typedef int i32x8 __attribute__((ext_vector_type(8)));

#define LDS_STRIDE 40   // ushorts per LDS row: 32 data + 4 dwords pad

#if __has_builtin(__builtin_amdgcn_tensor_load_to_lds) && \
    __has_builtin(__builtin_amdgcn_s_wait_tensorcnt)
#define TDM_OK 1
#else
#define TDM_OK 0
#endif

__device__ __forceinline__ unsigned short f32_to_bf16_bits(float f) {
  unsigned int u = __float_as_uint(f);
  u += 0x7FFFu + ((u >> 16) & 1u);          // round-to-nearest-even
  return (unsigned short)(u >> 16);
}

union FragU  { v16bf v; unsigned int u[8]; };
union Pack8U { uint4 q; unsigned short s[8]; };

// A-matrix 16x32 bf16 fragment; contiguous K (k = 16*lhalf + s), matching B.
__device__ __forceinline__ v16bf load_a_frag(const unsigned short* As, int mBase,
                                             int lrow, int lhalf) {
  FragU a;
  const unsigned int* arow = (const unsigned int*)(As + (mBase + lrow) * LDS_STRIDE);
#pragma unroll
  for (int j = 0; j < 8; ++j) {
    int kpair = 2 * j + 16 * lhalf;  // even
    a.u[j] = arow[kpair >> 1];
  }
  return a.v;
}

// B-matrix 32x16 bf16 fragment; Bs stored transposed [n][k] -> contiguous K.
__device__ __forceinline__ v16bf load_b_frag(const unsigned short* Bs, int nBase,
                                             int lrow, int lhalf) {
  FragU b;
  const unsigned int* brow =
      (const unsigned int*)(Bs + (nBase + lrow) * LDS_STRIDE + 16 * lhalf);
#pragma unroll
  for (int j = 0; j < 8; ++j) b.u[j] = brow[j];
  return b.v;
}

#if TDM_OK
// Issue one TDM load of a 32(K) x 128(M) bf16 tile into LDS, with LDS padding
// of 4 dwords every 16 dwords so rows land on the LDS_STRIDE=40 layout.
__device__ __forceinline__ void tdm_load_tile_a(const bf16_t* gsrc, unsigned ldsOff,
                                                int lda) {
  unsigned long long ga = (unsigned long long)(uintptr_t)gsrc;
  u32x4 g0;
  g0[0] = 1u;                                            // count=1, user mode
  g0[1] = ldsOff;                                        // lds_addr (bytes)
  g0[2] = (unsigned)(ga & 0xFFFFFFFFull);                // global_addr[31:0]
  g0[3] = (unsigned)((ga >> 32) & 0x01FFFFFFull) | (2u << 30);  // addr[56:32] | type=2
  i32x8 g1;
  // data_size=2B, pad_enable, pad_interval=16 dwords (enc 3), pad_amount=4 dwords (enc 3)
  g1[0] = (int)((1u << 16) | (1u << 20) | (3u << 22) | (3u << 25));
  unsigned td0 = (unsigned)lda;                          // tensor_dim0 (elements)
  unsigned td1 = 1u << 20;                               // tensor_dim1 (large)
  g1[1] = (int)((td0 & 0xFFFFu) << 16);                  // atomic_bar=0 | dim0 lo
  g1[2] = (int)(((td0 >> 16) & 0xFFFFu) | ((td1 & 0xFFFFu) << 16));
  g1[3] = (int)(((td1 >> 16) & 0xFFFFu) | (32u << 16));  // tile_dim0 = 32
  g1[4] = 128;                                           // tile_dim1 = 128, tile_dim2 = 0
  g1[5] = (int)td0;                                      // tensor_dim0_stride lo
  g1[6] = 0;
  g1[7] = 0;
  i32x4 gz = {0, 0, 0, 0};
#if __clang_major__ >= 23
  i32x8 gz8 = {0, 0, 0, 0, 0, 0, 0, 0};
  __builtin_amdgcn_tensor_load_to_lds(g0, g1, gz, gz, gz8, 0);
#else
  __builtin_amdgcn_tensor_load_to_lds(g0, g1, gz, gz, 0);
#endif
}
#endif

// ---------------------------------------------------------------------------
// Tiled GEMM: C[M,N] = A[M,K] @ B[K,N], bf16 in, bf16 WMMA, f32 accumulate.
// Block tile 128 x BN x 32; 256 threads = 8 waves; wave tile 32 x (BN/2).
// MODE 0: bf16 store               MODE 1: f32 store + bias[col]
// MODE 2: QKV epilogue (relu/scale + head-major bf16 scatter)
// MODE 3: per-head bf16 scatter (z=b*8+h) into [b, n, h*64+col]
// ---------------------------------------------------------------------------
template <int MODE, int BN>
__global__ __launch_bounds__(256)
void gemm_bf16_wmma(const bf16_t* __restrict__ A, const bf16_t* __restrict__ B,
                    float* __restrict__ Cf, unsigned short* __restrict__ Cb,
                    int K, int lda, int ldb, int ldc,
                    long long strideA, long long strideB,
                    const float* __restrict__ bias,
                    unsigned short* __restrict__ Qd, unsigned short* __restrict__ Kd,
                    unsigned short* __restrict__ Vd, int ntok, float qscale) {
  __shared__ unsigned short As[128 * LDS_STRIDE];
  __shared__ unsigned short Bs[BN * LDS_STRIDE];

  constexpr int NF = BN / 32;        // B fragments per wave

  const int tid   = threadIdx.x;
  const int wave  = tid >> 5;
  const int lane  = tid & 31;
  const int lrow  = lane & 15;
  const int lhalf = lane >> 4;

  const int blockN = blockIdx.x * BN;
  const int blockM = blockIdx.y * 128;
  const int z      = blockIdx.z;

  A += (long long)z * strideA;
  B += (long long)z * strideB;

  const int waveM = (wave & 3) * 32;
  const int waveN = (wave >> 2) * (BN / 2);

  v8f acc[2][NF];
#pragma unroll
  for (int i = 0; i < 2; ++i)
#pragma unroll
    for (int j = 0; j < NF; ++j)
#pragma unroll
      for (int e = 0; e < 8; ++e) acc[i][j][e] = 0.0f;

#if TDM_OK
  const unsigned asOff = (unsigned)(uintptr_t)(&As[0]);
#endif

  for (int k0 = 0; k0 < K; k0 += 32) {
    // ---- A tile 128x32 bf16 -> LDS ----
#if TDM_OK
    if (tid < 32) {                      // one TDM op per workgroup iteration
      tdm_load_tile_a(A + (long long)blockM * lda + k0, asOff, lda);
    }
#else
#pragma unroll
    for (int i = 0; i < 2; ++i) {
      int idx = tid + i * 256;           // 512 chunks of 8 bf16
      int m  = idx >> 2;
      int kc = (idx & 3) << 3;
      uint4 q = *(const uint4*)(A + (long long)(blockM + m) * lda + (k0 + kc));
      *(uint4*)&As[m * LDS_STRIDE + kc] = q;
    }
#endif
    // ---- B tile 32xBN bf16 -> LDS transposed [n][k] ----
#pragma unroll
    for (int i = 0; i < BN / 64; ++i) {
      int idx = tid + i * 256;           // 4*BN chunks of 8 bf16
      int k  = idx / (BN / 8);
      int nc = (idx % (BN / 8)) << 3;
      Pack8U p;
      p.q = *(const uint4*)(B + (long long)(k0 + k) * ldb + (blockN + nc));
      if (i == 0 && k0 + 32 < K)         // far prefetch of next B tile
        __builtin_prefetch(B + (long long)(k0 + 32 + k) * ldb + (blockN + nc), 0, 0);
#pragma unroll
      for (int e = 0; e < 8; ++e) Bs[(nc + e) * LDS_STRIDE + k] = p.s[e];
    }
#if TDM_OK
    if (tid < 32) __builtin_amdgcn_s_wait_tensorcnt(0);
#endif
    __syncthreads();

    v16bf a0 = load_a_frag(As, waveM,      lrow, lhalf);
    v16bf a1 = load_a_frag(As, waveM + 16, lrow, lhalf);
    v16bf bf[NF];
#pragma unroll
    for (int j = 0; j < NF; ++j) bf[j] = load_b_frag(Bs, waveN + 16 * j, lrow, lhalf);

#pragma unroll
    for (int j = 0; j < NF; ++j) {
      acc[0][j] = __builtin_amdgcn_wmma_f32_16x16x32_bf16(false, a0, false, bf[j],
                                                          (short)0, acc[0][j], false, false);
      acc[1][j] = __builtin_amdgcn_wmma_f32_16x16x32_bf16(false, a1, false, bf[j],
                                                          (short)0, acc[1][j], false, false);
    }
    __syncthreads();
  }

  // ---- epilogue: D layout m = g + 8*lhalf, n = lrow ----
#pragma unroll
  for (int i = 0; i < 2; ++i) {
#pragma unroll
    for (int j = 0; j < NF; ++j) {
#pragma unroll
      for (int g = 0; g < 8; ++g) {
        int row = blockM + waveM + i * 16 + g + 8 * lhalf;
        int col = blockN + waveN + j * 16 + lrow;
        float v = acc[i][j][g];
        if (MODE == 0) {
          Cb[(long long)row * ldc + col] = f32_to_bf16_bits(v);
        } else if (MODE == 1) {
          Cf[(long long)row * ldc + col] = v + bias[col];
        } else if (MODE == 2) {
          int b = row / ntok, nidx = row % ntok;
          int part = col / 512, rem = col % 512;
          int h = rem >> 6, dd = rem & 63;
          float o = v;
          if (part == 0) o = fmaxf(o, 0.0f) * qscale;      // q: relu * dh^-0.5
          else if (part == 1) o = fmaxf(o, 0.0f);          // k: relu
          unsigned short* dst = (part == 0) ? Qd : ((part == 1) ? Kd : Vd);
          dst[((long long)(b * 8 + h) * ntok + nidx) * 64 + dd] = f32_to_bf16_bits(o);
        } else {  // MODE 3
          int b = z >> 3, h = z & 7;
          Cb[((long long)(b * ntok + row) * 512) + h * 64 + col] = f32_to_bf16_bits(v);
        }
      }
    }
  }
}

// ---------------------------------------------------------------------------
// KV = K'^T @ V per head (64x64, f32 atomic split-K reduction over tokens).
// 128 threads = 4 waves (2x2 of 32x32). Inputs bf16, transposed into LDS.
// ---------------------------------------------------------------------------
__global__ __launch_bounds__(128)
void kv_outer_kernel(const bf16_t* __restrict__ Kp, const bf16_t* __restrict__ V,
                     float* __restrict__ KV, int ntok, int chunk) {
  __shared__ unsigned short As[64 * LDS_STRIDE];
  __shared__ unsigned short Bs[64 * LDS_STRIDE];

  const int tid   = threadIdx.x;
  const int wave  = tid >> 5;
  const int lane  = tid & 31;
  const int lrow  = lane & 15;
  const int lhalf = lane >> 4;

  const int z = blockIdx.x;                       // head index b*8+h
  const long long headA = (long long)z * ntok * 64;
  const int nStart = blockIdx.y * chunk;

  const int waveM = (wave & 1) * 32;
  const int waveN = (wave >> 1) * 32;

  v8f acc[2][2];
#pragma unroll
  for (int i = 0; i < 2; ++i)
#pragma unroll
    for (int j = 0; j < 2; ++j)
#pragma unroll
      for (int e = 0; e < 8; ++e) acc[i][j][e] = 0.0f;

  for (int kt = 0; kt < chunk; kt += 32) {
    int nBase = nStart + kt;
    // Stage 32 token rows (64 bf16 each) of K' and V, transposed so that
    // A[m=d][k=n_local] = K'[n][d] and B[n=d][k=n_local] = V[n][d].
#pragma unroll
    for (int i = 0; i < 2; ++i) {
      int idx = tid + i * 128;                    // 256 chunks of 8 bf16
      int nl  = idx >> 3;                         // token row 0..31
      int dc  = (idx & 7) << 3;                   // head-dim chunk 0..56
      Pack8U kq, vq;
      kq.q = *(const uint4*)(Kp + headA + (long long)(nBase + nl) * 64 + dc);
      vq.q = *(const uint4*)(V  + headA + (long long)(nBase + nl) * 64 + dc);
#pragma unroll
      for (int e = 0; e < 8; ++e) {
        As[(dc + e) * LDS_STRIDE + nl] = kq.s[e];
        Bs[(dc + e) * LDS_STRIDE + nl] = vq.s[e];
      }
    }
    __syncthreads();

    v16bf a0 = load_a_frag(As, waveM,      lrow, lhalf);
    v16bf a1 = load_a_frag(As, waveM + 16, lrow, lhalf);
    v16bf b0 = load_b_frag(Bs, waveN,      lrow, lhalf);
    v16bf b1 = load_b_frag(Bs, waveN + 16, lrow, lhalf);

    acc[0][0] = __builtin_amdgcn_wmma_f32_16x16x32_bf16(false, a0, false, b0,
                                                        (short)0, acc[0][0], false, false);
    acc[0][1] = __builtin_amdgcn_wmma_f32_16x16x32_bf16(false, a0, false, b1,
                                                        (short)0, acc[0][1], false, false);
    acc[1][0] = __builtin_amdgcn_wmma_f32_16x16x32_bf16(false, a1, false, b0,
                                                        (short)0, acc[1][0], false, false);
    acc[1][1] = __builtin_amdgcn_wmma_f32_16x16x32_bf16(false, a1, false, b1,
                                                        (short)0, acc[1][1], false, false);
    __syncthreads();
  }

  float* dst = KV + (long long)z * 4096;
#pragma unroll
  for (int i = 0; i < 2; ++i)
#pragma unroll
    for (int j = 0; j < 2; ++j)
#pragma unroll
      for (int g = 0; g < 8; ++g) {
        int row = waveM + i * 16 + g + 8 * lhalf;
        int col = waveN + j * 16 + lrow;
        atomicAdd(&dst[row * 64 + col], acc[i][j][g]);
      }
}

// ---------------------------------------------------------------------------
// Elementwise f32 -> bf16 (vectorized x4). n4 = element count / 4.
// ---------------------------------------------------------------------------
__global__ __launch_bounds__(256)
void cvt_f32_bf16(const float* __restrict__ s, unsigned short* __restrict__ d, int n4) {
  int i = blockIdx.x * 256 + threadIdx.x;
  if (i < n4) {
    float4 f = ((const float4*)s)[i];
    uint2 o;
    o.x = (unsigned)f32_to_bf16_bits(f.x) | ((unsigned)f32_to_bf16_bits(f.y) << 16);
    o.y = (unsigned)f32_to_bf16_bits(f.z) | ((unsigned)f32_to_bf16_bits(f.w) << 16);
    ((uint2*)d)[i] = o;
  }
}

// ---------------------------------------------------------------------------
extern "C" void kernel_launch(void* const* d_in, const int* in_sizes, int n_in,
                              void* d_out, int out_size, void* d_ws, size_t ws_size,
                              hipStream_t stream) {
  (void)in_sizes; (void)n_in; (void)out_size; (void)ws_size;
  const float* x    = (const float*)d_in[0];   // [4, 2048, 512]
  const float* Wqkv = (const float*)d_in[1];   // [512, 1536]
  const float* R    = (const float*)d_in[2];   // [64, 64]
  const float* Wout = (const float*)d_in[3];   // [512, 512]
  const float* bout = (const float*)d_in[4];   // [512]
  float* out = (float*)d_out;                  // [4, 2048, 512]

  const int Bb = 4, Ntok = 2048, H = 8, DH = 64;
  const int M1 = Bb * Ntok;                                   // 8192
  const long long headElems = (long long)Bb * H * Ntok * DH;  // 4,194,304

  // Workspace layout (bf16 unless noted).
  unsigned short* wsb = (unsigned short*)d_ws;
  unsigned short* xb     = wsb;                               // 4,194,304
  unsigned short* Wqkvb  = xb + 4194304;                      //   786,432
  unsigned short* Rb     = Wqkvb + 786432;                    //     4,096
  unsigned short* Woutb  = Rb + 4096;                         //   262,144
  unsigned short* Q      = Woutb + 262144;                    // head-major bf16
  unsigned short* Kt     = Q  + headElems;
  unsigned short* V      = Kt + headElems;
  unsigned short* Qp     = V  + headElems;
  unsigned short* Kp     = Qp + headElems;
  unsigned short* KVb    = Kp + headElems;                    //   131,072
  float*          KVf    = (float*)(KVb + 131072);            // f32 accumulators
  unsigned short* O      = Q;                                 // reuse (Q dead)

  const float qscale = 0.125f;      // 64^-0.5

  // 0) one-time f32 -> bf16 conversion of all GEMM operands
  cvt_f32_bf16<<<dim3((1048576 + 255) / 256), 256, 0, stream>>>(x,    xb,    1048576);
  cvt_f32_bf16<<<dim3((196608  + 255) / 256), 256, 0, stream>>>(Wqkv, Wqkvb, 196608);
  cvt_f32_bf16<<<dim3((1024    + 255) / 256), 256, 0, stream>>>(R,    Rb,    1024);
  cvt_f32_bf16<<<dim3((65536   + 255) / 256), 256, 0, stream>>>(Wout, Woutb, 65536);

  // 1) QKV = x @ Wqkv with fused relu/scale + head-major scatter (BN=128)
  dim3 g1(1536 / 128, M1 / 128, 1);
  gemm_bf16_wmma<2, 128><<<g1, 256, 0, stream>>>(
      (const bf16_t*)xb, (const bf16_t*)Wqkvb, nullptr, nullptr,
      512, 512, 1536, 0, 0, 0, nullptr, Q, Kt, V, Ntok, qscale);

  // 2) Q' = Q @ R ; K' = K @ R  (M = b*h*n, K = N = 64, BN=64)
  dim3 g2(1, (Bb * H * Ntok) / 128, 1);
  gemm_bf16_wmma<0, 64><<<g2, 256, 0, stream>>>(
      (const bf16_t*)Q, (const bf16_t*)Rb, nullptr, Qp,
      64, 64, 64, 64, 0, 0, nullptr, nullptr, nullptr, nullptr, Ntok, 0.0f);
  gemm_bf16_wmma<0, 64><<<g2, 256, 0, stream>>>(
      (const bf16_t*)Kt, (const bf16_t*)Rb, nullptr, Kp,
      64, 64, 64, 64, 0, 0, nullptr, nullptr, nullptr, nullptr, Ntok, 0.0f);

  // 3) KV = K'^T @ V per head, split-K (8 chunks of 256 tokens) + f32 atomics
  hipMemsetAsync(KVf, 0, (size_t)Bb * H * DH * DH * sizeof(float), stream);
  dim3 g3(Bb * H, Ntok / 256, 1);
  kv_outer_kernel<<<g3, 128, 0, stream>>>((const bf16_t*)Kp, (const bf16_t*)V,
                                          KVf, Ntok, 256);
  cvt_f32_bf16<<<dim3((32768 + 255) / 256), 256, 0, stream>>>(KVf, KVb, 32768);

  // 4) O = Q' @ KV per head, scattered to [b, n, h*64+d] bf16 (BN=64)
  dim3 g4(1, Ntok / 128, Bb * H);
  gemm_bf16_wmma<3, 64><<<g4, 256, 0, stream>>>(
      (const bf16_t*)Qp, (const bf16_t*)KVb, nullptr, O,
      64, 64, 64, 512, (long long)Ntok * 64, 4096, nullptr, nullptr, nullptr, nullptr,
      Ntok, 0.0f);

  // 5) out = O @ Wout + bout  (f32 output, BN=128)
  dim3 g5(512 / 128, M1 / 128, 1);
  gemm_bf16_wmma<1, 128><<<g5, 256, 0, stream>>>(
      (const bf16_t*)O, (const bf16_t*)Woutb, out, nullptr,
      512, 512, 512, 512, 0, 0, bout, nullptr, nullptr, nullptr, Ntok, 0.0f);
}
